// ResidualVectorQuantizer_83923660964603
// MI455X (gfx1250) — matrix operand order repaired and checked by reading, not compiled, not run on hardware
//
#include <hip/hip_runtime.h>
#include <hip/hip_bf16.h>

typedef __attribute__((ext_vector_type(16))) __bf16 v16bf;
typedef __attribute__((ext_vector_type(8)))  float  v8f;

#define N_ROWS  131072
#define DIM     128
#define NQ      8
#define NCODE   1024
#define ROWS_PER_BLOCK 128
#define WAVES   8
#define CT_CODES 128          // codes staged in LDS per tile
#define LDS_STRIDE 144        // bf16 elems per code row: 128 + 16 pad -> 288B (32B multiple)

__device__ __forceinline__ __bf16 f2bf(float f) { return (__bf16)f; }

__device__ __forceinline__ unsigned int pack2bf(float lo, float hi) {
    const unsigned short a = __builtin_bit_cast(unsigned short, (__bf16)lo);
    const unsigned short b = __builtin_bit_cast(unsigned short, (__bf16)hi);
    return (unsigned int)a | ((unsigned int)b << 16);
}

// d_out layout: [ z_q : N*DIM floats ][ loss : 1 float ][ indices-as-float : N*NQ ]
__global__ __launch_bounds__(256) void rvq_kernel(const float* __restrict__ z,
                                                  const float* __restrict__ cbs,
                                                  float* __restrict__ zq_out,
                                                  float* __restrict__ idx_out,
                                                  float* __restrict__ block_loss) {
    __shared__ __bf16 ldsB[CT_CODES * LDS_STRIDE];
    __shared__ float  cnormS[CT_CODES];
    __shared__ int    idxS[WAVES][16];
    __shared__ float  lossS[WAVES];

    const int tid  = threadIdx.x;
    const int wave = tid >> 5;
    const int lane = tid & 31;
    const int hf   = lane >> 4;     // which half of the wave
    const int m    = lane & 15;     // A-matrix row owned by this lane
    const int rowBase = blockIdx.x * ROWS_PER_BLOCK + wave * 16;
    const int myRow   = rowBase + m;

    // Residual in registers. Index i = c*8 + j maps to column
    //   k = c*16 + hf*8 + j
    // which matches the documented 16-bit A-fragment layout per K32 chunk.
    float res[64];
#pragma unroll
    for (int c = 0; c < 8; ++c) {
        const int kbase = c * 16 + hf * 8;
#pragma unroll
        for (int j = 0; j < 8; ++j)
            res[c * 8 + j] = z[(size_t)myRow * DIM + kbase + j];
    }

    float lsum = 0.0f;

    for (int qs = 0; qs < NQ; ++qs) {
        // Build A fragments (4 chunks of K=32) from the current residual.
        v16bf a[4];
#pragma unroll
        for (int kk = 0; kk < 4; ++kk)
#pragma unroll
            for (int e = 0; e < 16; ++e)
                a[kk][e] = f2bf(res[kk * 16 + e]);

        float mval[8];
        int   midx[8];
#pragma unroll
        for (int v = 0; v < 8; ++v) { mval[v] = 3.4e38f; midx[v] = 0; }

        const float* cq = cbs + (size_t)qs * NCODE * DIM;

        for (int ct = 0; ct < NCODE / CT_CODES; ++ct) {
            __syncthreads();   // protect ldsB reuse
            // Stage 128 codes (fp32 -> bf16) into LDS; 2 columns per thread
            // per iteration -> one packed ds_store_b32, coalesced b64 reads.
            const float* gsrc = cq + (size_t)ct * CT_CODES * DIM;
            for (int i = 0; i < (CT_CODES * DIM) / 512; ++i) {
                const int e  = i * 512 + tid * 2;
                const int r  = e >> 7;
                const int cc = e & 127;
                *(unsigned int*)&ldsB[r * LDS_STRIDE + cc] =
                    pack2bf(gsrc[e], gsrc[e + 1]);
            }
            __syncthreads();
            // Per-code squared norms (of the bf16-rounded codes, consistent with the WMMA dot).
            if (tid < CT_CODES) {
                float s = 0.0f;
                for (int k = 0; k < DIM; ++k) {
                    const float cv = (float)ldsB[tid * LDS_STRIDE + k];
                    s += cv * cv;
                }
                cnormS[tid] = s;
            }
            __syncthreads();

            // Two column tiles per iteration with the accumulator chains
            // interleaved: each dependent WMMA->WMMA is separated by the
            // sibling chain's WMMA, and the score/min VALU of tile 0 covers
            // the WMMA->VALU hazard of tile 1 (no v_nops needed).
            for (int nt = 0; nt < CT_CODES / 16; nt += 2) {
                const int n0 = nt * 16 + m;       // B column owned by this lane
                const int n1 = n0 + 16;
                v8f acc0 = {};
                v8f acc1 = {};
#pragma unroll
                for (int kk = 0; kk < 4; ++kk) {
                    // B fragment: element s holds K = (s>>3)*16 + hf*8 + (s&7) (matches A)
                    v16bf b0, b1;
#pragma unroll
                    for (int s = 0; s < 16; ++s) {
                        b0[s] = ldsB[n0 * LDS_STRIDE + kk * 32 + (s >> 3) * 16 + hf * 8 + (s & 7)];
                        b1[s] = ldsB[n1 * LDS_STRIDE + kk * 32 + (s >> 3) * 16 + hf * 8 + (s & 7)];
                    }
                    acc0 = __builtin_amdgcn_wmma_f32_16x16x32_bf16(
                        false, a[kk], false, b0, (short)0, acc0, false, false);
                    acc1 = __builtin_amdgcn_wmma_f32_16x16x32_bf16(
                        false, a[kk], false, b1, (short)0, acc1, false, false);
                }
                const float cn0 = cnormS[n0];
                const float cn1 = cnormS[n1];
                const int g0 = ct * CT_CODES + n0;
                const int g1 = g0 + 16;
#pragma unroll
                for (int v = 0; v < 8; ++v) {
                    const float s0 = cn0 - 2.0f * acc0[v];   // ||c||^2 - 2 r.c
                    if (s0 < mval[v]) { mval[v] = s0; midx[v] = g0; }
                }
#pragma unroll
                for (int v = 0; v < 8; ++v) {
                    const float s1 = cn1 - 2.0f * acc1[v];
                    if (s1 < mval[v]) { mval[v] = s1; midx[v] = g1; }
                }
            }
        }

        // Reduce argmin across the 16 lanes sharing this half (each saw a
        // distinct column residue). Tie-break: lowest index (matches argmin).
#pragma unroll
        for (int off = 1; off < 16; off <<= 1) {
#pragma unroll
            for (int v = 0; v < 8; ++v) {
                const float ov = __shfl_xor(mval[v], off, 32);
                const int   oi = __shfl_xor(midx[v], off, 32);
                if (ov < mval[v] || (ov == mval[v] && oi < midx[v])) {
                    mval[v] = ov; midx[v] = oi;
                }
            }
        }

        // C layout: VGPR v <-> row M = v + 8*hf. Lane with m==0 publishes.
        if (m == 0) {
#pragma unroll
            for (int v = 0; v < 8; ++v) {
                const int rr = hf * 8 + v;
                idxS[wave][rr] = midx[v];
                idx_out[(size_t)(rowBase + rr) * NQ + qs] = (float)midx[v];
            }
        }
        __syncthreads();

        // Gather selected code (fp32, L2-resident), update residual + loss.
        const int sel = idxS[wave][m];
        const float* qrow = cq + (size_t)sel * DIM;
#pragma unroll
        for (int c = 0; c < 8; ++c) {
            const int kbase = c * 16 + hf * 8;
#pragma unroll
            for (int j = 0; j < 8; ++j) {
                const float d = res[c * 8 + j] - qrow[kbase + j];
                lsum += d * d;
                res[c * 8 + j] = d;
            }
        }
    }

    // z_q = z - final residual
#pragma unroll
    for (int c = 0; c < 8; ++c) {
        const int kbase = c * 16 + hf * 8;
#pragma unroll
        for (int j = 0; j < 8; ++j)
            zq_out[(size_t)myRow * DIM + kbase + j] =
                z[(size_t)myRow * DIM + kbase + j] - res[c * 8 + j];
    }

    // Deterministic per-block loss partial.
#pragma unroll
    for (int off = 16; off >= 1; off >>= 1)
        lsum += __shfl_xor(lsum, off, 32);
    if (lane == 0) lossS[wave] = lsum;
    __syncthreads();
    if (tid == 0) {
        float t = 0.0f;
        for (int w = 0; w < WAVES; ++w) t += lossS[w];
        block_loss[blockIdx.x] = t;
    }
}

__global__ __launch_bounds__(256) void rvq_loss_reduce(const float* __restrict__ block_loss,
                                                       float* __restrict__ loss_out,
                                                       int nblocks) {
    __shared__ float s[256];
    float t = 0.0f;
    for (int i = threadIdx.x; i < nblocks; i += 256) t += block_loss[i];
    s[threadIdx.x] = t;
    __syncthreads();
    for (int o = 128; o > 0; o >>= 1) {
        if ((int)threadIdx.x < o) s[threadIdx.x] += s[threadIdx.x + o];
        __syncthreads();
    }
    if (threadIdx.x == 0)
        *loss_out = s[0] * (1.25f / ((float)NQ * (float)N_ROWS * (float)DIM));
}

extern "C" void kernel_launch(void* const* d_in, const int* in_sizes, int n_in,
                              void* d_out, int out_size, void* d_ws, size_t ws_size,
                              hipStream_t stream) {
    const float* z   = (const float*)d_in[0];   // [131072,128] fp32
    const float* cbs = (const float*)d_in[1];   // [8,1024,128] fp32
    float* out  = (float*)d_out;
    float* zq   = out;                                   // N*DIM
    float* loss = out + (size_t)N_ROWS * DIM;            // 1
    float* idxo = loss + 1;                              // N*NQ (indices as float)
    float* bl   = (float*)d_ws;                          // 1024 block partials

    const int nblocks = N_ROWS / ROWS_PER_BLOCK;         // 1024
    rvq_kernel<<<nblocks, 256, 0, stream>>>(z, cbs, zq, idxo, bl);
    rvq_loss_reduce<<<1, 256, 0, stream>>>(bl, loss, nblocks);
}
